// CRuleLoss_19877108646588
// MI455X (gfx1250) — matrix operand order repaired and checked by reading, not compiled, not run on hardware
//
#include <hip/hip_runtime.h>
#include <hip/hip_bf16.h>

// Problem constants (match reference)
#define BB   16384   // batch
#define CC   1000    // classes
#define KPAD 1024    // K padded to multiple of 32 (zero-filled)
#define NPAD 1008    // N padded to multiple of 16 (zero-filled H rows)

typedef __attribute__((ext_vector_type(16))) __bf16        v16bf;
typedef __attribute__((ext_vector_type(8)))  __bf16        v8bf;
typedef __attribute__((ext_vector_type(8)))  float         v8f;
typedef __attribute__((ext_vector_type(4)))  unsigned int  v4u;
typedef __attribute__((ext_vector_type(4)))  unsigned int  u32x4;
typedef __attribute__((ext_vector_type(8)))  int           i32x8;
typedef __attribute__((ext_vector_type(4)))  int           i32x4;

__device__ __forceinline__ unsigned short f32_to_bf16_rne(float f) {
    unsigned u = __builtin_bit_cast(unsigned, f);
    unsigned rounding = 0x7FFFu + ((u >> 16) & 1u);
    return (unsigned short)((u + rounding) >> 16);
}

__device__ __forceinline__ v8bf load_bf8(const unsigned short* p) {
    v4u raw = *reinterpret_cast<const v4u*>(p);   // 16B aligned by construction
    return __builtin_bit_cast(v8bf, raw);
}

__device__ __forceinline__ v16bf cat16(v8bf lo, v8bf hi) {
    return __builtin_shufflevector(lo, hi, 0,1,2,3,4,5,6,7,8,9,10,11,12,13,14,15);
}

// Issue a TDM 1-D copy of one contiguous 32 KB H panel (16 rows x KPAD bf16)
// from global memory into LDS.  D# layout per CDNA5 ISA.
// This toolchain exposes the 6-arg builtin:
//   (uint32x4 g0, int32x8 g1, int32x4 g2, int32x4 g3, int32x8 g4, i32 cpol)
__device__ __forceinline__ void tdm_load_panel(unsigned lds_off,
                                               unsigned long long gaddr) {
    u32x4 g0;
    g0[0] = 1u;                                            // count=1, valid, user mode
    g0[1] = lds_off;                                       // lds_addr (bytes)
    g0[2] = (unsigned)(gaddr & 0xFFFFFFFFull);             // global_addr[31:0]
    g0[3] = (unsigned)((gaddr >> 32) & 0x01FFFFFFull)      // global_addr[56:32]
          | (2u << 30);                                    // type=2 ("image")

    i32x8 g1;
    g1[0] = 0x00010000;        // workgroup_mask=0, data_size=1 (2 bytes)
    g1[1] = 0x40000000;        // atomic_barrier_addr=0 | tensor_dim0 lo16 (16384)<<16
    g1[2] = 0x00010000;        // tensor_dim0 hi16=0 | tensor_dim1=1 <<16
    g1[3] = 0x40000000;        // tensor_dim1 hi16=0 | tile_dim0=16384 <<16
    g1[4] = 0;                 // tile_dim1=0 (unused, 1-D tile), tile_dim2=0
    g1[5] = 16384;             // tensor_dim0_stride lo32
    g1[6] = 0;                 // stride hi bits / dim1_stride lo16
    g1[7] = 0;                 // dim1_stride hi32

    i32x4 z4 = {0, 0, 0, 0};               // groups 2/3 unused (<=2-D tensor)
    i32x8 z8 = {0, 0, 0, 0, 0, 0, 0, 0};   // extra group (clang-23 form), unused
    __builtin_amdgcn_tensor_load_to_lds(g0, g1, z4, z4, z8, 0);
}

// ---------------------------------------------------------------- kernels ---

__global__ void zero_card(float* card) {
    if (threadIdx.x == 0 && blockIdx.x == 0) card[0] = 0.0f;
}

// Y[f32, B x C]  ->  Yb[bf16, B x KPAD] (K zero-padded)
__global__ __launch_bounds__(256) void prep_y(const float* __restrict__ Yf,
                                              unsigned short* __restrict__ Yb) {
    size_t idx = (size_t)blockIdx.x * blockDim.x + threadIdx.x;
    if (idx >= (size_t)BB * KPAD) return;
    int b = (int)(idx >> 10);        // /KPAD
    int k = (int)(idx & (KPAD - 1));
    unsigned short v = 0;
    if (k < CC) v = f32_to_bf16_rne(Yf[(size_t)b * CC + k]);
    Yb[idx] = v;
}

// H[f32, C x C] -> Hb[bf16, NPAD x KPAD] (zero padded), plus cardinal = sum(H)
__global__ __launch_bounds__(256) void prep_h(const float* __restrict__ Hf,
                                              unsigned short* __restrict__ Hb,
                                              float* __restrict__ card) {
    size_t idx = (size_t)blockIdx.x * blockDim.x + threadIdx.x;
    float v = 0.0f;
    if (idx < (size_t)NPAD * KPAD) {
        int n = (int)(idx >> 10);
        int k = (int)(idx & (KPAD - 1));
        if (n < CC && k < CC) v = Hf[(size_t)n * CC + k];
        Hb[idx] = f32_to_bf16_rne(v);
    }
    // wave32 reduction of v, one atomic per wave
    float s = v;
    #pragma unroll
    for (int off = 16; off >= 1; off >>= 1) s += __shfl_xor(s, off, 32);
    if ((threadIdx.x & 31) == 0 && s != 0.0f) atomicAdd(card, s);
}

// Fused GEMM + per-row reductions.
// Each wave owns 16 batch rows; all 8 waves in the block share the B operand
// (an H panel) which is TDM-prefetched into double-buffered LDS.  For every
// 16-col N tile, accumulate a 16x16 WMMA tile over K=KPAD, then fold it into
// rsum (= sum_n Z) and wsum (= sum_n Y*Z) without ever materializing Z.
__global__ __launch_bounds__(256) void gemm_loss(const unsigned short* __restrict__ Yb,
                                                 const unsigned short* __restrict__ Hb,
                                                 const float* __restrict__ Yf,
                                                 const float* __restrict__ card,
                                                 float* __restrict__ out) {
    __shared__ unsigned short smem[2][16 * KPAD];   // 2 x 32 KB H panels

    const int lane = threadIdx.x & 31;
    const int wave = threadIdx.x >> 5;
    const int tile = blockIdx.x * 8 + wave;     // 0..1023 row tiles
    const int b0   = tile * 16;

    const int half = lane >> 4;    // 0: lanes 0-15, 1: lanes 16-31
    const int lid  = lane & 15;

    // Flat pointer low 32 bits == LDS byte offset (ISA §10.2 aperture rule)
    const unsigned lds0 = (unsigned)(unsigned long long)(uintptr_t)&smem[0][0];
    const unsigned lds1 = (unsigned)(unsigned long long)(uintptr_t)&smem[1][0];

    // A fragment source: row M = lid, K chunks {8*half .. +7} and {+16 .. +23}
    const unsigned short* Arow = Yb + (size_t)(b0 + lid) * KPAD + 8 * half;

    float rsum[8], wsum[8];
    #pragma unroll
    for (int i = 0; i < 8; ++i) { rsum[i] = 0.0f; wsum[i] = 0.0f; }

    const int NT = NPAD / 16;   // 63 N tiles

    // Prologue: TDM-load panel 0 into buffer 0 (wave 0 only; TDM ignores EXEC)
    if (wave == 0) {
        tdm_load_panel(lds0, (unsigned long long)(uintptr_t)Hb);
    }

    for (int t = 0; t < NT; ++t) {
        const int cur = t & 1;

        if (wave == 0) __builtin_amdgcn_s_wait_tensorcnt(0);  // panel t arrived
        __syncthreads();   // publish panel t; all readers of the other buffer done

        if (wave == 0 && (t + 1) < NT) {   // prefetch panel t+1 into other buffer
            tdm_load_panel(cur ? lds0 : lds1,
                           (unsigned long long)(uintptr_t)
                               (Hb + (size_t)(t + 1) * 16 * KPAD));
        }

        // B fragment source (LDS): col n = t*16+lid, K chunks {8*half .. +7} and {+16 .. +23}
        const unsigned short* Brow = &smem[cur][(size_t)lid * KPAD + 8 * half];

        v8f acc = {};
        #pragma unroll 4
        for (int k0 = 0; k0 < KPAD; k0 += 32) {
            v16bf a = cat16(load_bf8(Arow + k0), load_bf8(Arow + k0 + 16));
            v16bf b = cat16(load_bf8(Brow + k0), load_bf8(Brow + k0 + 16));
            acc = __builtin_amdgcn_wmma_f32_16x16x32_bf16(
                false, a, false, b, (short)0, acc, false, false);
        }

        // C/D layout: acc[i] = Z[b0 + i + 8*half, t*16 + lid]
        const int col = t * 16 + lid;
        if (col < CC) {
            const float* yp = Yf + col + (size_t)(b0 + 8 * half) * CC;
            #pragma unroll
            for (int i = 0; i < 8; ++i) {
                float yv = yp[(size_t)i * CC];
                rsum[i] += acc[i];
                wsum[i] += acc[i] * yv;
            }
        }
        // padded H rows are zero -> acc == 0 there, nothing to add
    }

    // Reduce across the 16 lanes of each half (xor masks < 16 stay in-half)
    #pragma unroll
    for (int i = 0; i < 8; ++i) {
        float r = rsum[i], w = wsum[i];
        #pragma unroll
        for (int off = 8; off >= 1; off >>= 1) {
            r += __shfl_xor(r, off, 32);
            w += __shfl_xor(w, off, 32);
        }
        rsum[i] = r; wsum[i] = w;
    }

    const float inv = 1.0f / card[0];
    const int rowbase = b0 + 8 * half;
    #pragma unroll
    for (int i = 0; i < 8; ++i) {
        if (lid == i) out[rowbase + i] = (rsum[i] - wsum[i]) * inv;
    }
}

// ---------------------------------------------------------------- launcher --

extern "C" void kernel_launch(void* const* d_in, const int* in_sizes, int n_in,
                              void* d_out, int out_size, void* d_ws, size_t ws_size,
                              hipStream_t stream) {
    const float* Yf = (const float*)d_in[0];   // y_pred [B, C] f32
    const float* Hf = (const float*)d_in[1];   // H      [C, C] f32
    // d_in[2] = y_true (int64) -- unused by the reference math
    float* out = (float*)d_out;                // [B] f32

    // Workspace layout:
    //   [0, 4)                      : float cardinal
    //   [256, 256 + B*KPAD*2)       : Y in bf16, K padded      (33.55 MB)
    //   [.., + NPAD*KPAD*2)         : H in bf16, zero padded   ( 2.06 MB)
    char* ws = (char*)d_ws;
    float*          card = (float*)ws;
    unsigned short* Yb   = (unsigned short*)(ws + 256);
    unsigned short* Hb   = (unsigned short*)(ws + 256 + (size_t)BB * KPAD * 2);

    zero_card<<<1, 1, 0, stream>>>(card);

    {
        size_t n = (size_t)BB * KPAD;
        prep_y<<<(unsigned)((n + 255) / 256), 256, 0, stream>>>(Yf, Yb);
    }
    {
        size_t n = (size_t)NPAD * KPAD;
        prep_h<<<(unsigned)((n + 255) / 256), 256, 0, stream>>>(Hf, Hb, card);
    }

    // 1024 row tiles of 16 rows, 8 waves (one tile each) per 256-thread block
    gemm_loss<<<BB / 16 / 8, 256, 0, stream>>>(Yb, Hb, Yf, card, out);
}
